// JointNetwork_83648783057727
// MI455X (gfx1250) — matrix operand order repaired and checked by reading, not compiled, not run on hardware
//
#include <hip/hip_runtime.h>
#include <hip/hip_bf16.h>

// RNN-T joint network for MI455X (gfx1250), wave32 + WMMA bf16.
// Pipeline: pack weights -> enc/pred projections (WMMA) -> fused tanh+vocab GEMM (WMMA).
// Inner GEMM loops: rolled (unroll 1) 2-step rotation => batched s_clause loads with
// partial s_wait_loadcnt under the WMMA chain (distance-1 prefetch kept by the scheduler).

typedef __attribute__((ext_vector_type(16))) __bf16 v16bf;
typedef __attribute__((ext_vector_type(8)))  float  v8f;

struct Pair32 { uint4 lo; uint4 hi; };   // 32 bytes == one v16bf A-fragment

__device__ __forceinline__ unsigned short f2bf(float x) {
  // round-to-nearest-even fp32 -> bf16 (inputs finite)
  unsigned int u = __float_as_uint(x);
  u += 0x7FFFu + ((u >> 16) & 1u);
  return (unsigned short)(u >> 16);
}
__device__ __forceinline__ unsigned int pk2bf(float lo, float hi) {
  return (unsigned int)f2bf(lo) | ((unsigned int)f2bf(hi) << 16);
}

// A-fragment from LDS: two 16B runs per lane (K = k0..k0+7, k0+16..k0+23)
__device__ __forceinline__ v16bf loadA(const unsigned short* p_) {
  const uint4* p = (const uint4*)p_;
  Pair32 fr; fr.lo = p[0]; fr.hi = p[2];
  return __builtin_bit_cast(v16bf, fr);
}
// NTPW B-fragments, contiguous packed tiles (512 ushorts each), 32B per lane
template <int NTPW>
__device__ __forceinline__ void loadB(const unsigned short* base, int laneOff, v16bf* b) {
#pragma unroll
  for (int a = 0; a < NTPW; ++a)
    b[a] = *(const v16bf*)(base + a * 512 + laneOff);
}
template <int NTPW>
__device__ __forceinline__ void mmaChain(v16bf a, const v16bf* b, v8f* acc) {
#pragma unroll
  for (int i = 0; i < NTPW; ++i)
    acc[i] = __builtin_amdgcn_wmma_f32_16x16x32_bf16(
        false, a, false, b[i], (short)0, acc[i], false, false);
}

// ---------------------------------------------------------------------------
// Pack W[K][N] (fp32, row-major) into bf16 B-matrix fragment order:
//   out[((ks*NT + nt)*32 + lane)*16 + e] = bf16( W[(ks*32 + 16*(e/8) + 8*(lane/16) + e%8)*N + nt*16 + lane%16] )
// ---------------------------------------------------------------------------
__global__ __launch_bounds__(256)
void pack_w_kernel(const float* __restrict__ W, unsigned short* __restrict__ out,
                   int K, int N) {
  int gid = blockIdx.x * 256 + threadIdx.x;
  if (gid >= K * N) return;
  int e    = gid & 15;
  int lane = (gid >> 4) & 31;
  int rest = gid >> 9;            // = ks*NT + nt
  int NT   = N >> 4;
  int nt   = rest % NT;
  int ks   = rest / NT;
  int k = ks * 32 + ((e >> 3) << 4) + ((lane >> 4) << 3) + (e & 7);
  int n = nt * 16 + (lane & 15);
  out[gid] = f2bf(W[k * N + n]);
}

// ---------------------------------------------------------------------------
// Projection GEMM: out[M][N] = A[M][K] @ W + bias, fp32 in/out, bf16 WMMA core.
// 256 threads = 8 waves (2 M-waves x 4 N-waves), NTPW n-tiles per wave.
// WG tile: 32 rows x (4*16*NTPW) cols -- divides N exactly, no guards.
// K, N compile-time (immediate-offset addressing); Ksteps = K/32 even.
// ---------------------------------------------------------------------------
template <int NTPW, int K, int N>
__global__ __launch_bounds__(256)
void proj_gemm_kernel(const float* __restrict__ A, const unsigned short* __restrict__ Wp,
                      const float* __restrict__ bias, float* __restrict__ out) {
  __shared__ unsigned short ldsA[32 * K];
  const int tid  = threadIdx.x;
  const int lane = tid & 31;
  const int w    = __builtin_amdgcn_readfirstlane(tid >> 5);   // wave-uniform -> SGPR
  const int m0   = blockIdx.x * 32;

  {  // stage 32 contiguous rows as bf16 (float4 reads -> packed dword pairs)
    const float4* A4 = (const float4*)(A + (size_t)m0 * K);
    const int n4 = (32 * K) >> 2;
    for (int i = tid; i < n4; i += 256) {
      float4 v = A4[i];
      uint2 pk; pk.x = pk2bf(v.x, v.y); pk.y = pk2bf(v.z, v.w);
      *(uint2*)&ldsA[i * 4] = pk;
    }
  }
  __syncthreads();

  constexpr int NT = N >> 4;
  constexpr int Ksteps = K >> 5;                // even
  const int mwave = w >> 2;                     // 0..1
  const int nwave = w & 3;                      // 0..3
  const int nt0  = (blockIdx.y * 4 + nwave) * NTPW;
  const int kb   = (lane < 16) ? 0 : 8;
  const int laneOff = lane * 16;                // ushorts

  v8f acc[NTPW];
#pragma unroll
  for (int a = 0; a < NTPW; ++a) {              // bias folded into acc init
    float bv = bias[(nt0 + a) * 16 + (lane & 15)];
#pragma unroll
    for (int v = 0; v < 8; ++v) acc[a][v] = bv;
  }

  const unsigned short* ap = &ldsA[(mwave * 16 + (lane & 15)) * K + kb];
  const unsigned short* wb = Wp + (size_t)nt0 * 512;
  constexpr size_t wstep = (size_t)NT * 512;    // ushorts per k-step

  // rolled software pipeline, distance-1 prefetch, 2-step rotation
  v16bf a0, a1, b0[NTPW], b1[NTPW];
  a0 = loadA(ap);
  loadB<NTPW>(wb, laneOff, b0);
#pragma unroll 1
  for (int ks = 0; ks + 2 < Ksteps; ks += 2) {
    a1 = loadA(ap + (ks + 1) * 32);
    loadB<NTPW>(wb + (size_t)(ks + 1) * wstep, laneOff, b1);
    mmaChain<NTPW>(a0, b0, acc);
    a0 = loadA(ap + (ks + 2) * 32);
    loadB<NTPW>(wb + (size_t)(ks + 2) * wstep, laneOff, b0);
    mmaChain<NTPW>(a1, b1, acc);
  }
  a1 = loadA(ap + (Ksteps - 1) * 32);
  loadB<NTPW>(wb + (size_t)(Ksteps - 1) * wstep, laneOff, b1);
  mmaChain<NTPW>(a0, b0, acc);
  mmaChain<NTPW>(a1, b1, acc);

  const int mr = m0 + mwave * 16 + ((lane < 16) ? 0 : 8);
#pragma unroll
  for (int a = 0; a < NTPW; ++a) {
    int n = (nt0 + a) * 16 + (lane & 15);
#pragma unroll
    for (int v = 0; v < 8; ++v)
      out[(size_t)(mr + v) * N + n] = acc[a][v];
  }
}

// ---------------------------------------------------------------------------
// Fused joint kernel: rows m = b*T*U + t*U + u (B=4,T=256,U=64), K=640, V=1024.
//   joint = tanh(encp[b,t,:] + predp[b,u,:])  (bf16 in LDS)
//   logits = clamp(joint @ W_out + b_out, +/-15)   (non-temporal stores)
// WG tile 32x256 (8 waves = 2M x 4N, 4 n-tiles/wave), grid (2048, 4).
// ---------------------------------------------------------------------------
__global__ __launch_bounds__(256)
void joint_gemm_kernel(const float* __restrict__ encp, const float* __restrict__ predp,
                       const unsigned short* __restrict__ Wp, const float* __restrict__ bias,
                       float* __restrict__ out) {
  __shared__ unsigned short ldsJ[32 * 640];     // 40 KB
  const int tid  = threadIdx.x;
  const int lane = tid & 31;
  const int w    = __builtin_amdgcn_readfirstlane(tid >> 5);
  const int m0   = blockIdx.x * 32;

  // stage tanh(enc+pred) rows as bf16
  for (int i = tid; i < (32 * 640) >> 2; i += 256) {
    int idx = i * 4;
    int r = idx / 640, k = idx - r * 640;
    int m = m0 + r;
    int b = m >> 14;                            // /(T*U)
    int t = (m >> 6) & 255;
    int u = m & 63;
    float4 e4 = *(const float4*)&encp[(size_t)(b * 256 + t) * 640 + k];
    float4 p4 = *(const float4*)&predp[(size_t)(b * 64 + u) * 640 + k];
    uint2 pk;
    pk.x = pk2bf(tanhf(e4.x + p4.x), tanhf(e4.y + p4.y));
    pk.y = pk2bf(tanhf(e4.z + p4.z), tanhf(e4.w + p4.w));
    *(uint2*)&ldsJ[idx] = pk;
  }
  __syncthreads();

  const int mwave = w >> 2, nwave = w & 3;
  const int nt0  = (blockIdx.y * 4 + nwave) * 4;   // NT = 64 total
  const int kb   = (lane < 16) ? 0 : 8;
  const int laneOff = lane * 16;

  v8f acc[4];
#pragma unroll
  for (int a = 0; a < 4; ++a) {
    float bv = bias[(nt0 + a) * 16 + (lane & 15)];
#pragma unroll
    for (int v = 0; v < 8; ++v) acc[a][v] = bv;
  }

  const unsigned short* ap = &ldsJ[(mwave * 16 + (lane & 15)) * 640 + kb];
  const unsigned short* wb = Wp + (size_t)nt0 * 512;
  constexpr size_t wstep = (size_t)64 * 512;    // NT=64 tiles per k-step

  // rolled software pipeline, distance-1 prefetch, 2-step rotation (Ksteps = 20)
  v16bf a0, a1, b0[4], b1[4];
  a0 = loadA(ap);
  loadB<4>(wb, laneOff, b0);
#pragma unroll 1
  for (int ks = 0; ks + 2 < 20; ks += 2) {
    a1 = loadA(ap + (ks + 1) * 32);
    loadB<4>(wb + (size_t)(ks + 1) * wstep, laneOff, b1);
    mmaChain<4>(a0, b0, acc);
    a0 = loadA(ap + (ks + 2) * 32);
    loadB<4>(wb + (size_t)(ks + 2) * wstep, laneOff, b0);
    mmaChain<4>(a1, b1, acc);
  }
  a1 = loadA(ap + 19 * 32);
  loadB<4>(wb + (size_t)19 * wstep, laneOff, b1);
  mmaChain<4>(a0, b0, acc);
  mmaChain<4>(a1, b1, acc);

  const int mr = m0 + mwave * 16 + ((lane < 16) ? 0 : 8);
#pragma unroll
  for (int a = 0; a < 4; ++a) {
    int n = (nt0 + a) * 16 + (lane & 15);
#pragma unroll
    for (int v = 0; v < 8; ++v) {
      float x = fminf(fmaxf(acc[a][v], -15.0f), 15.0f);
      __builtin_nontemporal_store(x, &out[(size_t)(mr + v) * 1024 + n]);
    }
  }
}

// ---------------------------------------------------------------------------
extern "C" void kernel_launch(void* const* d_in, const int* in_sizes, int n_in,
                              void* d_out, int out_size, void* d_ws, size_t ws_size,
                              hipStream_t stream) {
  const float* enc    = (const float*)d_in[0];   // (4,256,512)
  const float* pred   = (const float*)d_in[1];   // (4,64,640)
  const float* W_enc  = (const float*)d_in[2];   // (512,640)
  const float* b_enc  = (const float*)d_in[3];   // (640)
  const float* W_pred = (const float*)d_in[4];   // (640,640)
  const float* b_pred = (const float*)d_in[5];   // (640)
  const float* W_out  = (const float*)d_in[6];   // (640,1024)
  const float* b_out  = (const float*)d_in[7];   // (1024)
  float* logits = (float*)d_out;                 // (4,256,64,1024)

  char* ws = (char*)d_ws;                        // all offsets 256B aligned
  unsigned short* Wenc_p  = (unsigned short*)(ws + 0);        //  512*640*2 = 655360
  unsigned short* Wpred_p = (unsigned short*)(ws + 655360);   //  640*640*2 = 819200
  unsigned short* Wout_p  = (unsigned short*)(ws + 1474560);  //  640*1024*2 = 1310720
  float*          encp    = (float*)(ws + 2785280);           // 1024*640*4 = 2621440
  float*          predp   = (float*)(ws + 5406720);           //  256*640*4 = 655360
                                                              // total ~5.9 MB

  // 1) pack weights into bf16 WMMA B-fragment layout
  pack_w_kernel<<<(512 * 640 + 255) / 256, 256, 0, stream>>>(W_enc,  Wenc_p,  512, 640);
  pack_w_kernel<<<(640 * 640 + 255) / 256, 256, 0, stream>>>(W_pred, Wpred_p, 640, 640);
  pack_w_kernel<<<(640 * 1024 + 255) / 256, 256, 0, stream>>>(W_out, Wout_p, 640, 1024);

  // 2) projections: NTPW=5 -> 4 N-waves x 5 tiles x 16 = 320 cols/WG, grid.y=2 covers N=640.
  proj_gemm_kernel<5, 512, 640><<<dim3(1024 / 32, 2), 256, 0, stream>>>(enc,  Wenc_p,  b_enc,  encp);
  proj_gemm_kernel<5, 640, 640><<<dim3(256 / 32, 2),  256, 0, stream>>>(pred, Wpred_p, b_pred, predp);

  // 3) fused tanh-broadcast + vocab GEMM + bias + clamp
  joint_gemm_kernel<<<dim3(65536 / 32, 4), 256, 0, stream>>>(encp, predp, Wout_p, b_out, logits);
}
